// EfficientSelfAttention_22608707846573
// MI455X (gfx1250) — compile-verified, matrix operand order checked by reading, not run on hardware
//
#include <hip/hip_runtime.h>

// ---------------------------------------------------------------------------
// EfficientSelfAttention (PVT-style SRA) for MI455X / gfx1250, wave32 + WMMA.
// B=8, N=16384 (128x128), C=64, NH=1, HD=64, R=8 -> M=256.
// All GEMMs via v_wmma_f32_16x16x32_bf16 (bf16 inputs, f32 accumulate).
// All WMMA operands stored K-fast (A as [M][K], B as B^T = [N][K]) so each
// bf16 fragment load is exactly 2x ds_load_b128 per lane (f32 ones 4x).
// Global->LDS staging uses GLOBAL_LOAD_ASYNC_TO_LDS_B128 (ASYNCcnt) when the
// toolchain exposes the gfx1250 async builtins, else falls back to b128 copy.
// ---------------------------------------------------------------------------

typedef __attribute__((ext_vector_type(16))) __bf16 v16bf;
typedef __attribute__((ext_vector_type(8)))  __bf16 bf16x8;
typedef __attribute__((ext_vector_type(8)))  float  v8f;
typedef __attribute__((ext_vector_type(4)))  float  f32x4;
typedef __attribute__((ext_vector_type(4)))  unsigned int u32x4;
typedef __attribute__((ext_vector_type(4)))  int    i32x4;

#define C64   64
#define MTOT  256
#define NTOT  16384
#define IMGW  128
#define KCONV 4096   // 64 pixels * 64 channels per patch

#if defined(__gfx1250__) && \
    __has_builtin(__builtin_amdgcn_global_load_async_to_lds_b128) && \
    __has_builtin(__builtin_amdgcn_s_wait_asynccnt)
#define HAVE_ASYNC_LDS 1
// Builtin signature (from hipcc diagnostic): (int4 AS1*, int4 AS3*, Ii, Ii)
typedef __attribute__((address_space(1))) i32x4 gi32x4;
typedef __attribute__((address_space(3))) i32x4 li32x4;
#else
#define HAVE_ASYNC_LDS 0
#endif

// 16-byte global->LDS copy: async (no VGPR round-trip, ASYNCcnt) if available.
__device__ __forceinline__ void cp16(void* dst_lds, const void* src_glb) {
#if HAVE_ASYNC_LDS
  __builtin_amdgcn_global_load_async_to_lds_b128(
      (gi32x4*)src_glb, (li32x4*)dst_lds, /*offset=*/0, /*cpol=*/0);
#else
  *(u32x4*)dst_lds = *(const u32x4*)src_glb;
#endif
}

__device__ __forceinline__ void cp_commit() {
#if HAVE_ASYNC_LDS
  __builtin_amdgcn_s_wait_asynccnt(0);
#endif
}

// --- WMMA fragment loaders (CDNA5 16-bit layouts, wave32) -------------------
// A 16x32 (MxK): lane r=lane&15 holds row r; K = 8h+0..7 (VGPR0-3) and
// 16+8h+0..7 (VGPR4-7), h = lane>>4.  B 32x16 (KxN) is identical with the
// lane index meaning column N, reading from B^T stored [N][K].
// C/D 16x16 f32: VGPR i -> row i+8h, col = lane&15.

__device__ __forceinline__ v16bf frag_rm_bf16(const __bf16* __restrict__ src, int stride) {
  const int lane = threadIdx.x & 31;
  const int r = lane & 15, h = lane >> 4;
  const __bf16* p = src + r * stride + h * 8;
  bf16x8 lo = *(const bf16x8*)(p);        // K = 8h + 0..7      (ds_load_b128)
  bf16x8 hi = *(const bf16x8*)(p + 16);   // K = 16 + 8h + 0..7 (ds_load_b128)
  return __builtin_shufflevector(lo, hi, 0, 1, 2, 3, 4, 5, 6, 7,
                                 8, 9, 10, 11, 12, 13, 14, 15);
}

__device__ __forceinline__ v16bf frag_rm_f32(const float* __restrict__ src, int stride) {
  const int lane = threadIdx.x & 31;
  const int r = lane & 15, h = lane >> 4;
  const float* p = src + r * stride + h * 8;
  f32x4 f0 = *(const f32x4*)(p);
  f32x4 f1 = *(const f32x4*)(p + 4);
  f32x4 f2 = *(const f32x4*)(p + 16);
  f32x4 f3 = *(const f32x4*)(p + 20);
  v16bf a;
#pragma unroll
  for (int i = 0; i < 4; ++i) {
    a[i]      = (__bf16)f0[i];
    a[4 + i]  = (__bf16)f1[i];
    a[8 + i]  = (__bf16)f2[i];
    a[12 + i] = (__bf16)f3[i];
  }
  return a;
}

__device__ __forceinline__ v8f wmma_bf16(v16bf a, v16bf b, v8f c) {
  return __builtin_amdgcn_wmma_f32_16x16x32_bf16(
      /*neg_a=*/false, a, /*neg_b=*/false, b,
      /*c_mod=*/(short)0, c, /*reuse_a=*/false, /*reuse_b=*/false);
}

// ---------------------------------------------------------------------------
// Kernel 0: weight preprocessing -> bf16, B^T ([N][K], K-fast) layouts.
//   qwB  [64][64]   = bf16(q_w)        (q_w is already [out][in] = [N][K])
//   pwB  [64][64]   = bf16(proj_w)
//   kvwB [128][64]  = bf16(kv_w)
//   convBT[64][4096]: convBT[o][p*64+i] = sr_w[o][i][dy][dx], p = dy*8+dx
// ---------------------------------------------------------------------------
__global__ __launch_bounds__(256)
void prep_kernel(const float* __restrict__ q_w, const float* __restrict__ kv_w,
                 const float* __restrict__ proj_w, const float* __restrict__ sr_w,
                 __bf16* __restrict__ qwB, __bf16* __restrict__ kvwB,
                 __bf16* __restrict__ pwB, __bf16* __restrict__ convBT) {
  const int idx0 = blockIdx.x * blockDim.x + threadIdx.x;
  const int stride = gridDim.x * blockDim.x;
  for (int idx = idx0; idx < 64 * 64; idx += stride) {
    qwB[idx] = (__bf16)q_w[idx];
    pwB[idx] = (__bf16)proj_w[idx];
  }
  for (int idx = idx0; idx < 128 * 64; idx += stride) {
    kvwB[idx] = (__bf16)kv_w[idx];
  }
  for (int idx = idx0; idx < 64 * KCONV; idx += stride) {
    const int o = idx >> 12, k = idx & 4095;    // k = p*64 + i
    const int p = k >> 6, i = k & 63;
    convBT[idx] = (__bf16)sr_w[o * KCONV + i * 64 + p];
  }
}

// ---------------------------------------------------------------------------
// Kernel 1: spatial reduction path.
// One workgroup (4 waves / 128 thr) per 16 output positions (m-rows).
// conv-as-GEMM ([16 x 4096] @ [4096 x 64], one 8x8-patch pixel per iter) ->
// +bias -> LayerNorm(64) -> kv GEMM ([16 x 64] @ [64 x 128]) ->
// store kM [B][256][64] (scale 1/8 folded) and vT [B][64][256], bf16.
// ---------------------------------------------------------------------------
__global__ __launch_bounds__(128)
void kv_kernel(const float* __restrict__ x, const float* __restrict__ sr_b,
               const float* __restrict__ ln_w, const float* __restrict__ ln_b,
               const float* __restrict__ kv_b,
               const __bf16* __restrict__ convBT, const __bf16* __restrict__ kvwB,
               __bf16* __restrict__ kM, __bf16* __restrict__ vT) {
  __shared__ __attribute__((aligned(16))) float  sA[16 * 64];    // patch pixels f32
  __shared__ __attribute__((aligned(16))) __bf16 sB[64 * 64];    // convBT chunk
  __shared__ __attribute__((aligned(16))) __bf16 sKVW[128 * 64]; // kv_w bf16
  __shared__ __attribute__((aligned(16))) float  sXR[16 * 64];   // conv out pre-LN
  __shared__ __attribute__((aligned(16))) __bf16 sXLN[16 * 64];  // post-LN bf16

  const int b = blockIdx.y, mt = blockIdx.x;
  const int tid = threadIdx.x, wave = tid >> 5, lane = tid & 31;
  const int col16 = lane & 15, h = lane >> 4;

  // stage kv weights once (16 KB = 1024 x 16B, exactly 8 per thread)
#pragma unroll
  for (int it = 0; it < 8; ++it) {
    const int i = tid + it * 128;
    cp16((u32x4*)sKVW + i, (const u32x4*)kvwB + i);
  }

  // ---- conv as GEMM: one patch pixel (K=64) per iteration ----
  v8f acc = {0.f, 0.f, 0.f, 0.f, 0.f, 0.f, 0.f, 0.f};
  for (int p = 0; p < 64; ++p) {
    const int dy = p >> 3, dx = p & 7;
    __syncthreads();
    // A: 16 m-rows x 64 channels (contiguous f32) = 256 x 16B, 2 per thread
#pragma unroll
    for (int it = 0; it < 2; ++it) {
      const int t = tid + it * 128;
      const int r = t >> 4, j = t & 15;
      const int m = mt * 16 + r;
      const int ty = m >> 4, tx = m & 15;
      const int n = (ty * 8 + dy) * IMGW + (tx * 8 + dx);
      cp16(sA + r * 64 + j * 4, x + ((size_t)b * NTOT + n) * C64 + j * 4);
    }
    // B: convBT columns [p*64, p*64+64) for 64 out-ch rows = 512 x 16B
#pragma unroll
    for (int it = 0; it < 4; ++it) {
      const int t = tid + it * 128;
      const int o = t >> 3, q = t & 7;
      cp16(sB + o * 64 + q * 8, convBT + (size_t)o * KCONV + p * 64 + q * 8);
    }
    if (p + 1 < 64)  // prefetch next pixel's B row 0 (global_prefetch path)
      __builtin_prefetch(convBT + (size_t)(p + 1) * 64, 0, 1);
    cp_commit();
    __syncthreads();
#pragma unroll
    for (int kc = 0; kc < 2; ++kc) {
      v16bf af = frag_rm_f32(sA + kc * 32, 64);
      v16bf bf = frag_rm_bf16(sB + wave * 16 * 64 + kc * 32, 64);
      acc = wmma_bf16(af, bf, acc);
    }
  }
  __syncthreads();

  {  // +bias, stash for LN
    const int col = wave * 16 + col16;
    const float bias = sr_b[col];
#pragma unroll
    for (int i = 0; i < 8; ++i) sXR[(i + 8 * h) * 64 + col] = acc[i] + bias;
  }
  __syncthreads();

  if (tid < 16) {  // LayerNorm over 64 channels, one thread per m-row
    float mu = 0.f, m2 = 0.f;
    for (int cc = 0; cc < 64; ++cc) {
      const float v0 = sXR[tid * 64 + cc];
      mu += v0; m2 += v0 * v0;
    }
    mu *= (1.f / 64.f); m2 *= (1.f / 64.f);
    const float rs = rsqrtf(m2 - mu * mu + 1e-5f);
    for (int cc = 0; cc < 64; ++cc) {
      const float v0 = (sXR[tid * 64 + cc] - mu) * rs * ln_w[cc] + ln_b[cc];
      sXLN[tid * 64 + cc] = (__bf16)v0;
    }
  }
  __syncthreads();

  // ---- kv GEMM: [16 x 64] @ kv_w^T; wave handles out-cols [32w, 32w+32) ----
#pragma unroll
  for (int t2 = 0; t2 < 2; ++t2) {
    const int ncol = wave * 32 + t2 * 16;
    v8f a2 = {0.f, 0.f, 0.f, 0.f, 0.f, 0.f, 0.f, 0.f};
#pragma unroll
    for (int kc = 0; kc < 2; ++kc) {
      v16bf af = frag_rm_bf16(sXLN + kc * 32, 64);
      v16bf bf = frag_rm_bf16(sKVW + ncol * 64 + kc * 32, 64);  // B^T rows
      a2 = wmma_bf16(af, bf, a2);
    }
    const int col = ncol + col16;
    const float bias = kv_b[col];
#pragma unroll
    for (int i = 0; i < 8; ++i) {
      const int m = mt * 16 + i + 8 * h;
      const float v0 = a2[i] + bias;
      if (col < 64)  // k natural [B][256][64] (= B^T for scores), scale folded
        kM[((size_t)b * MTOT + m) * 64 + col] = (__bf16)(v0 * 0.125f);
      else           // v transposed [B][64][256] (= B^T for attn@v)
        vT[((size_t)b * 64 + (col - 64)) * MTOT + m] = (__bf16)v0;
    }
  }
}

// ---------------------------------------------------------------------------
// Kernel 2: fused q-proj + attention + output projection.
// One workgroup (8 waves / 256 thr) per 32 rows of N; k/v fully LDS-resident.
// Dynamic LDS = 144 KB (fits easily in CDNA5's 320 KB / WGP).
// ---------------------------------------------------------------------------
__global__ __launch_bounds__(256)
void attn_kernel(const float* __restrict__ x, const float* __restrict__ q_b,
                 const float* __restrict__ proj_b,
                 const __bf16* __restrict__ kM, const __bf16* __restrict__ vT,
                 const __bf16* __restrict__ qwB, const __bf16* __restrict__ pwB,
                 float* __restrict__ out) {
  extern __shared__ __align__(16) char smem[];
  __bf16* sKB = (__bf16*)smem;       // [256][64]  32 KB  k (B^T for scores)
  __bf16* sVT = sKB + MTOT * 64;     // [64][256]  32 KB  v^T (B^T for attn@v)
  __bf16* sQW = sVT + 64 * MTOT;     // [64][64]    8 KB
  __bf16* sPW = sQW + 64 * 64;       // [64][64]    8 KB
  __bf16* sQ  = sPW + 64 * 64;       // [32][64]    4 KB
  __bf16* sP  = sQ + 32 * 64;        // [32][256]  16 KB
  __bf16* sO  = sP + 32 * MTOT;      // [32][64]    4 KB
  float*  sX  = (float*)(sO + 32 * 64);  // [32][64]   8 KB
  float*  sS  = sX + 32 * 64;            // [32][256] 32 KB

  const int b = blockIdx.y;
  const int n0 = blockIdx.x * 32;
  const int tid = threadIdx.x, wave = tid >> 5, lane = tid & 31;
  const int col16 = lane & 15, h = lane >> 4;

  {  // stage k, v^T, weights, x-tile (16B granules, exact trip counts)
    const u32x4* s1 = (const u32x4*)(kM + (size_t)b * MTOT * 64);
#pragma unroll
    for (int it = 0; it < 8; ++it) {
      const int i = tid + it * 256;
      cp16((u32x4*)sKB + i, s1 + i);
    }
    const u32x4* s2 = (const u32x4*)(vT + (size_t)b * 64 * MTOT);
#pragma unroll
    for (int it = 0; it < 8; ++it) {
      const int i = tid + it * 256;
      cp16((u32x4*)sVT + i, s2 + i);
    }
#pragma unroll
    for (int it = 0; it < 2; ++it) {
      const int i = tid + it * 256;
      cp16((u32x4*)sQW + i, (const u32x4*)qwB + i);
      cp16((u32x4*)sPW + i, (const u32x4*)pwB + i);
      cp16((f32x4*)sX + i,
           (const f32x4*)(x + ((size_t)b * NTOT + n0) * C64) + i);
    }
    cp_commit();
  }
  __syncthreads();

  const int rg  = wave >> 2;  // row-group of 16 (0..1)
  const int cg4 = wave & 3;   // col-group of 16 within 64 (0..3)

  // ---- 1) q = x @ q_w^T + q_b -> sQ (bf16) ----
  {
    v8f a = {0.f, 0.f, 0.f, 0.f, 0.f, 0.f, 0.f, 0.f};
#pragma unroll
    for (int kc = 0; kc < 2; ++kc) {
      v16bf af = frag_rm_f32(sX + rg * 16 * 64 + kc * 32, 64);
      v16bf bf = frag_rm_bf16(sQW + cg4 * 16 * 64 + kc * 32, 64);
      a = wmma_bf16(af, bf, a);
    }
    const float bias = q_b[cg4 * 16 + col16];
#pragma unroll
    for (int i = 0; i < 8; ++i)
      sQ[(rg * 16 + i + 8 * h) * 64 + cg4 * 16 + col16] = (__bf16)(a[i] + bias);
  }
  __syncthreads();

  // ---- 2) scores = q @ k^T (scale pre-folded into k) -> sS (f32) ----
#pragma unroll
  for (int j = 0; j < 4; ++j) {
    const int cg = cg4 * 4 + j;  // 0..15 over M=256
    v8f a = {0.f, 0.f, 0.f, 0.f, 0.f, 0.f, 0.f, 0.f};
#pragma unroll
    for (int kc = 0; kc < 2; ++kc) {
      v16bf af = frag_rm_bf16(sQ + rg * 16 * 64 + kc * 32, 64);
      v16bf bf = frag_rm_bf16(sKB + cg * 16 * 64 + kc * 32, 64);
      a = wmma_bf16(af, bf, a);
    }
#pragma unroll
    for (int i = 0; i < 8; ++i)
      sS[(rg * 16 + i + 8 * h) * MTOT + cg * 16 + col16] = a[i];
  }
  __syncthreads();

  // ---- 3) softmax over M per row: 8 lanes per row, shfl_xor reductions ----
  {
    const int row = tid >> 3, seg = tid & 7;   // 32 rows x 8 segments of 32
    float* rowp = sS + row * MTOT + seg * 32;
    float mx = -1e30f;
#pragma unroll
    for (int j = 0; j < 32; ++j) mx = fmaxf(mx, rowp[j]);
#pragma unroll
    for (int d = 1; d < 8; d <<= 1) mx = fmaxf(mx, __shfl_xor(mx, d, 32));
    float sum = 0.f;
#pragma unroll
    for (int j = 0; j < 32; ++j) {
      const float e = __expf(rowp[j] - mx);
      rowp[j] = e;
      sum += e;
    }
#pragma unroll
    for (int d = 1; d < 8; d <<= 1) sum += __shfl_xor(sum, d, 32);
    const float inv = 1.f / sum;
    __bf16* pp = sP + row * MTOT + seg * 32;
#pragma unroll
    for (int j = 0; j < 32; ++j) pp[j] = (__bf16)(rowp[j] * inv);
  }
  __syncthreads();

  // ---- 4) out = P @ V -> sO (bf16) ----
  {
    v8f a = {0.f, 0.f, 0.f, 0.f, 0.f, 0.f, 0.f, 0.f};
#pragma unroll
    for (int kc = 0; kc < 8; ++kc) {
      v16bf af = frag_rm_bf16(sP + rg * 16 * MTOT + kc * 32, MTOT);
      v16bf bf = frag_rm_bf16(sVT + cg4 * 16 * MTOT + kc * 32, MTOT);
      a = wmma_bf16(af, bf, a);
    }
#pragma unroll
    for (int i = 0; i < 8; ++i)
      sO[(rg * 16 + i + 8 * h) * 64 + cg4 * 16 + col16] = (__bf16)a[i];
  }
  __syncthreads();

  // ---- 5) final = out @ proj_w^T + proj_b -> global (f32) ----
  {
    v8f a = {0.f, 0.f, 0.f, 0.f, 0.f, 0.f, 0.f, 0.f};
#pragma unroll
    for (int kc = 0; kc < 2; ++kc) {
      v16bf af = frag_rm_bf16(sO + rg * 16 * 64 + kc * 32, 64);
      v16bf bf = frag_rm_bf16(sPW + cg4 * 16 * 64 + kc * 32, 64);
      a = wmma_bf16(af, bf, a);
    }
    const float bias = proj_b[cg4 * 16 + col16];
#pragma unroll
    for (int i = 0; i < 8; ++i)
      out[((size_t)b * NTOT + n0 + rg * 16 + i + 8 * h) * C64 + cg4 * 16 + col16] =
          a[i] + bias;
  }
}

// ---------------------------------------------------------------------------
// Host-side launcher.  Workspace layout (bf16):
//   kM    [8][256][64]   262144 B   (k, natural layout, scale folded)
//   vT    [8][64][256]   262144 B   (v transposed)
//   qwB   [64][64]         8192 B
//   pwB   [64][64]         8192 B
//   kvwB  [128][64]       16384 B
//   convBT[64][4096]     524288 B
//   total ~1.03 MB
// ---------------------------------------------------------------------------
extern "C" void kernel_launch(void* const* d_in, const int* in_sizes, int n_in,
                              void* d_out, int out_size, void* d_ws, size_t ws_size,
                              hipStream_t stream) {
  const float* x      = (const float*)d_in[0];
  // d_in[1]=H, d_in[2]=W (128, hardcoded)
  const float* q_w    = (const float*)d_in[3];
  const float* q_b    = (const float*)d_in[4];
  const float* kv_w   = (const float*)d_in[5];
  const float* kv_b   = (const float*)d_in[6];
  const float* proj_w = (const float*)d_in[7];
  const float* proj_b = (const float*)d_in[8];
  const float* sr_w   = (const float*)d_in[9];
  const float* sr_b   = (const float*)d_in[10];
  const float* ln_w   = (const float*)d_in[11];
  const float* ln_b   = (const float*)d_in[12];
  float* out = (float*)d_out;

  char* ws = (char*)d_ws;
  __bf16* kM     = (__bf16*)(ws);
  __bf16* vT     = (__bf16*)(ws + 262144);
  __bf16* qwB    = (__bf16*)(ws + 524288);
  __bf16* pwB    = (__bf16*)(ws + 532480);
  __bf16* kvwB   = (__bf16*)(ws + 540672);
  __bf16* convBT = (__bf16*)(ws + 557056);

  prep_kernel<<<dim3(1024), 256, 0, stream>>>(q_w, kv_w, proj_w, sr_w,
                                              qwB, kvwB, pwB, convBT);

  kv_kernel<<<dim3(16, 8), 128, 0, stream>>>(x, sr_b, ln_w, ln_b, kv_b,
                                             convBT, kvwB, kM, vT);

  const size_t smem = 147456;  // 144 KB dynamic LDS
  attn_kernel<<<dim3(512, 8), 256, smem, stream>>>(x, q_b, proj_b, kM, vT,
                                                   qwB, pwB, out);
}
